// SpinTransformerModule_2473901162858
// MI455X (gfx1250) — matrix operand order repaired and checked by reading, not compiled, not run on hardware
//
#include <hip/hip_runtime.h>

// SpinTransformer fixed-point module for MI455X (gfx1250, wave32, WMMA).
// All GEMMs via v_wmma_f32_16x16x32_f16 (f16 in, f32 accumulate); master m kept f32.
// Hot loops double-buffered; sched_barrier pins next-step loads ahead of
// current-step WMMAs so loadcnt waits become partial instead of full drains.

typedef __attribute__((ext_vector_type(16))) _Float16 v16h;
typedef __attribute__((ext_vector_type(8)))  _Float16 v8h;
typedef __attribute__((ext_vector_type(8)))  float    v8f;

#define NTOK  1024
#define NHEAD 16
#define DHEAD 64
#define RSQ   31.0f
#define RVAL  5.567764362830022f
#define BETA_F 1.0f
#define NITER 20

#if __has_builtin(__builtin_amdgcn_sched_barrier)
#define SCHED_FENCE() __builtin_amdgcn_sched_barrier(0)
#else
#define SCHED_FENCE() asm volatile("" ::: "memory")
#endif

__device__ __forceinline__ v8f wmma16(v16h a, v16h b, v8f c) {
  return __builtin_amdgcn_wmma_f32_16x16x32_f16(false, a, false, b, (short)0, c, false, false);
}
__device__ __forceinline__ v16h cat8(v8h a, v8h b) {
  return __builtin_shufflevector(a, b, 0,1,2,3,4,5,6,7,8,9,10,11,12,13,14,15);
}
__device__ __forceinline__ void ds_wait() {
  asm volatile("s_wait_dscnt 0" ::: "memory");
}

// ---------------- W_qk -> f16 ----------------
__global__ void k_w_to_f16(const float* __restrict__ w, _Float16* __restrict__ w16, int n) {
  int i = blockIdx.x * blockDim.x + threadIdx.x;
  if (i < n) w16[i] = (_Float16)w[i];
}

// ------- normalize x per (head,row) to radius R; m_init = phi(x + 0.125) -------
// (J @ m0 == 0.125 componentwise since softmax rows sum to 1)
__global__ void k_norm_init(const float* __restrict__ x,
                            float* __restrict__ xh, _Float16* __restrict__ xn16,
                            float* __restrict__ m32, _Float16* __restrict__ m16,
                            _Float16* __restrict__ mT16, float* __restrict__ msq) {
  int gid = blockIdx.x * blockDim.x + threadIdx.x;   // 16384 = 16 heads * 1024 rows
  int h = gid >> 10, i = gid & 1023;
  const float* xr = x + (size_t)i * 1024 + h * 64;
  float ss = 0.f;
  #pragma unroll
  for (int d = 0; d < 64; ++d) { float v = xr[d]; ss += v * v; }
  float sc = RVAL / sqrtf(ss);
  size_t bh = ((size_t)h * 1024 + i) * 64;
  float fsq = 0.f;
  #pragma unroll
  for (int d = 0; d < 64; ++d) {
    float xv = sc * xr[d];
    xh[bh + d] = xv;
    xn16[(size_t)i * 1024 + h * 64 + d] = (_Float16)xv;
    float fv = xv + 0.125f;
    fsq += fv * fv;
  }
  float gf = sqrtf(1.f + fsq * (1.f / RSQ));
  float ms = 1.f / (1.f + gf);
  msq[h * 1024 + i] = ms * ms * fsq;
  #pragma unroll
  for (int d = 0; d < 64; ++d) {
    float mn = ms * (sc * xr[d] + 0.125f);
    m32[bh + d] = mn;
    m16[bh + d] = (_Float16)mn;
    mT16[((size_t)h * 64 + d) * 1024 + i] = (_Float16)mn;
  }
}

// ---------------- qk = xn @ W^T : one wave = 16 rows x 64 out cols ----------------
__global__ void __launch_bounds__(128) k_qk(const _Float16* __restrict__ xn16,
                                            const _Float16* __restrict__ w16,
                                            _Float16* __restrict__ q16,
                                            _Float16* __restrict__ k16) {
  int lane = threadIdx.x & 31, wave = threadIdx.x >> 5;
  int l16 = lane & 15, lhi = lane >> 4;
  int wid = blockIdx.x * 4 + wave;       // 0..2047
  int i0 = (wid >> 5) << 4;              // 64 i-tiles
  int o0 = (wid & 31) << 6;              // 32 o-tiles of 64
  v8f acc[4] = {};
  v16h An, Bn0, Bn1, Bn2, Bn3;
  auto loadk = [&](int k0) {
    const _Float16* ar = xn16 + (size_t)(i0 + l16) * 1024 + k0 + 16 * lhi;
    An = *(const v16h*)ar;
    const _Float16* br = w16 + (size_t)(o0 + l16) * 1024 + k0 + 16 * lhi;
    Bn0 = *(const v16h*)(br);
    Bn1 = *(const v16h*)(br + 16 * 1024);
    Bn2 = *(const v16h*)(br + 32 * 1024);
    Bn3 = *(const v16h*)(br + 48 * 1024);
  };
  loadk(0);
  for (int k0 = 0; k0 < 1024; k0 += 32) {
    v16h A = An, B0 = Bn0, B1 = Bn1, B2 = Bn2, B3 = Bn3;
    loadk((k0 + 32) & 1023);             // wrapped redundant load on last step
    SCHED_FENCE();                       // loads above must issue before WMMAs below
    acc[0] = wmma16(A, B0, acc[0]);
    acc[1] = wmma16(A, B1, acc[1]);
    acc[2] = wmma16(A, B2, acc[2]);
    acc[3] = wmma16(A, B3, acc[3]);
    SCHED_FENCE();
  }
  #pragma unroll
  for (int c = 0; c < 4; ++c) {
    #pragma unroll
    for (int v = 0; v < 8; ++v) {
      int row = i0 + v + 8 * lhi;
      int o = o0 + 16 * c + l16;
      _Float16 hv = (_Float16)acc[c][v];
      if (o < 1024) q16[((size_t)(o >> 6) * 1024 + row) * 64 + (o & 63)] = hv;
      else {
        int o2 = o - 1024;
        k16[((size_t)(o2 >> 6) * 1024 + row) * 64 + (o2 & 63)] = hv;
      }
    }
  }
}

// ------- J = softmax(q k^T), flash-style two-pass; also rowsum of J^2 -------
__global__ void __launch_bounds__(128) k_attn(const _Float16* __restrict__ q16,
                                              const _Float16* __restrict__ k16,
                                              _Float16* __restrict__ J16,
                                              float* __restrict__ rs) {
  __shared__ __attribute__((aligned(32))) float redm[4][16 * 16];
  __shared__ __attribute__((aligned(32))) float reds[4][16 * 16];
  __shared__ __attribute__((aligned(32))) float rstat[4][32];
  int lane = threadIdx.x & 31, wave = threadIdx.x >> 5;
  int l16 = lane & 15, lhi = lane >> 4;
  int wid = blockIdx.x * 4 + wave;       // 0..1023
  int h = wid >> 6;
  int i0 = (wid & 63) << 4;
  const _Float16* aq = q16 + ((size_t)h * 1024 + i0 + l16) * 64;
  v16h Aq0 = *(const v16h*)(aq + 16 * lhi);
  v16h Aq1 = *(const v16h*)(aq + 32 + 16 * lhi);
  v16h Bn0, Bn1;
  auto loadj = [&](int j0) {
    const _Float16* kr = k16 + ((size_t)h * 1024 + j0 + l16) * 64 + 16 * lhi;
    Bn0 = *(const v16h*)(kr);
    Bn1 = *(const v16h*)(kr + 32);
  };
  float mx[8], sm[8];
  #pragma unroll
  for (int v = 0; v < 8; ++v) { mx[v] = -3.0e38f; sm[v] = 0.f; }
  loadj(0);
  for (int j0 = 0; j0 < 1024; j0 += 16) {            // pass 1: online max/sum
    v16h B0 = Bn0, B1 = Bn1;
    loadj((j0 + 16) & 1023);
    SCHED_FENCE();
    v8f s = {};
    s = wmma16(Aq0, B0, s);
    s = wmma16(Aq1, B1, s);
    #pragma unroll
    for (int v = 0; v < 8; ++v) {
      float e = s[v];
      float nm = fmaxf(mx[v], e);
      sm[v] = sm[v] * __expf(mx[v] - nm) + __expf(e - nm);
      mx[v] = nm;
    }
  }
  #pragma unroll
  for (int v = 0; v < 8; ++v) {
    int row = v + 8 * lhi;
    redm[wave][row * 16 + l16] = mx[v];
    reds[wave][row * 16 + l16] = sm[v];
  }
  ds_wait();
  if (lane < 16) {
    float rm = -3.0e38f, rsum = 0.f;
    #pragma unroll
    for (int c = 0; c < 16; ++c) {
      float m_ = redm[wave][lane * 16 + c], s_ = reds[wave][lane * 16 + c];
      float nm = fmaxf(rm, m_);
      rsum = rsum * __expf(rm - nm) + s_ * __expf(m_ - nm);
      rm = nm;
    }
    rstat[wave][lane * 2]     = rm;
    rstat[wave][lane * 2 + 1] = 1.f / rsum;
  }
  ds_wait();
  float rq[8] = {};
  loadj(0);
  for (int j0 = 0; j0 < 1024; j0 += 16) {            // pass 2: write J (f16), sum J^2
    v16h B0 = Bn0, B1 = Bn1;
    loadj((j0 + 16) & 1023);
    SCHED_FENCE();
    v8f s = {};
    s = wmma16(Aq0, B0, s);
    s = wmma16(Aq1, B1, s);
    #pragma unroll
    for (int v = 0; v < 8; ++v) {
      int row = v + 8 * lhi;
      float p = __expf(s[v] - rstat[wave][row * 2]) * rstat[wave][row * 2 + 1];
      _Float16 ph = (_Float16)p;
      J16[((size_t)h << 20) + (size_t)(i0 + row) * 1024 + j0 + l16] = ph;
      float pf = (float)ph;
      rq[v] += pf * pf;
    }
  }
  #pragma unroll
  for (int v = 0; v < 8; ++v) { int row = v + 8 * lhi; redm[wave][row * 16 + l16] = rq[v]; }
  ds_wait();
  if (lane < 16) {
    float t = 0.f;
    #pragma unroll
    for (int c = 0; c < 16; ++c) t += redm[wave][lane * 16 + c];
    rs[h * 1024 + i0 + lane] = t;
  }
}

// ------- fused fixed-point iteration: one wave = 16 rows of one head -------
// per j-step of 32: Jm (4 wmma), S = m_I m_J^T (4 wmma), elementwise J^2 terms,
// W = J^2*S via LDS relayout, D += W@m (4 wmma). Then per-row scalar chain + phi.
__global__ void __launch_bounds__(128) k_iter(
    const _Float16* __restrict__ J16, const float* __restrict__ rs,
    const float* __restrict__ xh,
    const float* __restrict__ m32s, const _Float16* __restrict__ m16s,
    const _Float16* __restrict__ mT16s, const float* __restrict__ msqs,
    float* __restrict__ m32d, _Float16* __restrict__ m16d,
    _Float16* __restrict__ mT16d, float* __restrict__ msqd) {
  __shared__ __attribute__((aligned(32))) _Float16 jlds[4][16 * 32];
  __shared__ __attribute__((aligned(32))) _Float16 wlds[4][16 * 32];
  __shared__ __attribute__((aligned(32))) float jmlds[4][16 * 64];
  __shared__ __attribute__((aligned(32))) float dlds[4][16 * 64];
  __shared__ __attribute__((aligned(32))) float blds[4][16 * 16];
  __shared__ __attribute__((aligned(32))) float elds[4][16 * 16];
  int lane = threadIdx.x & 31, wave = threadIdx.x >> 5;
  int l16 = lane & 15, lhi = lane >> 4;
  int wid = blockIdx.x * 4 + wave;       // 0..1023
  int h = wid >> 6;
  int i0 = (wid & 63) << 4;
  int kb = 16 * lhi;
  // A operand: rows i0..i0+15 of m (K = d), reused across the whole j-loop
  const _Float16* mi = m16s + ((size_t)h * 1024 + i0 + l16) * 64;
  v16h Am0 = cat8(*(const v8h*)(mi + kb),      *(const v8h*)(mi + kb + 8));
  v16h Am1 = cat8(*(const v8h*)(mi + 32 + kb), *(const v8h*)(mi + 32 + kb + 8));
  v8f accJm[4] = {};
  v8f accD[4]  = {};
  float bAcc[8] = {};
  float eAcc[8] = {};
  const _Float16* jbase = J16 + ((size_t)h << 20) + (size_t)(i0 + l16) * 1024;
  // double-buffered tile registers
  v8h ja0n, ja1n;
  v16h Bmn0, Bmn1, Bmn2, Bmn3;
  v16h Bs00n, Bs01n, Bs10n, Bs11n;
  float msq0n, msq1n;
  auto loadstep = [&](int j0) {
    const _Float16* jr = jbase + j0;
    ja0n = *(const v8h*)(jr + kb);
    ja1n = *(const v8h*)(jr + kb + 8);
    const _Float16* bt = mT16s + ((size_t)h * 64 + l16) * 1024 + j0 + 16 * lhi;
    Bmn0 = *(const v16h*)(bt);
    Bmn1 = *(const v16h*)(bt + 16 * 1024);
    Bmn2 = *(const v16h*)(bt + 32 * 1024);
    Bmn3 = *(const v16h*)(bt + 48 * 1024);
    const _Float16* mj0 = m16s + ((size_t)h * 1024 + j0 + l16) * 64 + 16 * lhi;
    Bs00n = *(const v16h*)(mj0);
    Bs01n = *(const v16h*)(mj0 + 32);
    Bs10n = *(const v16h*)(mj0 + 16 * 64);
    Bs11n = *(const v16h*)(mj0 + 16 * 64 + 32);
    msq0n = msqs[h * 1024 + j0 + l16];
    msq1n = msqs[h * 1024 + j0 + 16 + l16];
  };
  loadstep(0);
  for (int j0 = 0; j0 < 1024; j0 += 32) {
    // rotate double buffer, then issue next-step loads before consuming current
    v8h ja0 = ja0n, ja1 = ja1n;
    v16h Bm0 = Bmn0, Bm1 = Bmn1, Bm2 = Bmn2, Bm3 = Bmn3;
    v16h Bs00 = Bs00n, Bs01 = Bs01n, Bs10 = Bs10n, Bs11 = Bs11n;
    float msq0 = msq0n, msq1 = msq1n;
    __builtin_prefetch((const void*)(jbase + ((j0 + 64) & 1023)), 0, 0);
    loadstep((j0 + 32) & 1023);
    SCHED_FENCE();                       // next-step loads issued before current WMMAs
    v16h AJ = cat8(ja0, ja1);
    *(v8h*)&jlds[wave][l16 * 32 + kb]      = ja0;
    *(v8h*)&jlds[wave][l16 * 32 + kb + 8] = ja1;
    // Jm accumulation (A = J tile, B = m^T strips)
    accJm[0] = wmma16(AJ, Bm0, accJm[0]);
    accJm[1] = wmma16(AJ, Bm1, accJm[1]);
    accJm[2] = wmma16(AJ, Bm2, accJm[2]);
    accJm[3] = wmma16(AJ, Bm3, accJm[3]);
    // S = m_I . m_J  (two 16x16 tiles over this 32-wide j strip, K = d = 64)
    v8f S0 = {}, S1 = {};
    S0 = wmma16(Am0, Bs00, S0);
    S0 = wmma16(Am1, Bs01, S0);
    S1 = wmma16(Am0, Bs10, S1);
    S1 = wmma16(Am1, Bs11, S1);
    ds_wait();
    // elementwise in C layout: B += J^2 S^2 ; E += J^2 msq_j ; W = J^2 S -> LDS
    #pragma unroll
    for (int s = 0; s < 2; ++s) {
      int jc = 16 * s + l16;
      float msqj = (s == 0) ? msq0 : msq1;
      #pragma unroll
      for (int v = 0; v < 8; ++v) {
        int row = v + 8 * lhi;
        float jv = (float)jlds[wave][row * 32 + jc];
        float jsq = jv * jv;
        float sv = (s == 0) ? S0[v] : S1[v];
        bAcc[v] += jsq * sv * sv;
        eAcc[v] += jsq * msqj;
        wlds[wave][row * 32 + jc] = (_Float16)(jsq * sv);
      }
    }
    ds_wait();
    v16h AW = cat8(*(const v8h*)&wlds[wave][l16 * 32 + kb],
                   *(const v8h*)&wlds[wave][l16 * 32 + kb + 8]);
    accD[0] = wmma16(AW, Bm0, accD[0]);
    accD[1] = wmma16(AW, Bm1, accD[1]);
    accD[2] = wmma16(AW, Bm2, accD[2]);
    accD[3] = wmma16(AW, Bm3, accD[3]);
    SCHED_FENCE();
  }
  // dump per-row reductions to LDS
  #pragma unroll
  for (int c = 0; c < 4; ++c) {
    #pragma unroll
    for (int v = 0; v < 8; ++v) {
      int row = v + 8 * lhi, col = 16 * c + l16;
      jmlds[wave][row * 64 + col] = accJm[c][v];
      dlds[wave][row * 64 + col]  = accD[c][v];
    }
  }
  #pragma unroll
  for (int v = 0; v < 8; ++v) {
    int row = v + 8 * lhi;
    blds[wave][row * 16 + l16] = bAcc[v];
    elds[wave][row * 16 + l16] = eAcc[v];
  }
  ds_wait();
  // per-row scalar chain (reference _f + phi), f32 master precision
  if (lane < 16) {
    int i = i0 + lane;
    size_t base = ((size_t)h * 1024 + i) * 64;
    const float* xr = xh + base;
    const float* mr = m32s + base;
    float msqv = 0.f;
    for (int d = 0; d < 64; ++d) msqv += mr[d] * mr[d];
    float alpha = 2.f * RSQ / (BETA_F * (RSQ - msqv));        // inv_phi scale
    float g = sqrtf(1.f + (BETA_F * BETA_F) * alpha * alpha * msqv / RSQ);
    float mv = 0.f, vsq = 0.f;
    for (int d = 0; d < 64; ++d) {
      float vd = xr[d] + jmlds[wave][lane * 64 + d] - alpha * mr[d];
      mv += mr[d] * vd;
      vsq += vd * vd;
    }
    float Bi = 0.f, Ei = 0.f;
    #pragma unroll
    for (int c = 0; c < 16; ++c) { Bi += blds[wave][lane * 16 + c]; Ei += elds[wave][lane * 16 + c]; }
    float rsv = rs[h * 1024 + i];
    float Ci = rsv * RSQ - Ei;                                 // Jsq @ (R^2 - msq)
    float og = 1.f / (1.f + g);
    float c1 = (BETA_F * BETA_F) * (1.f + 3.f * g) / (RSQ * RSQ * g * g * g);
    float c2 = (BETA_F * BETA_F) / (RSQ * (g * g + g));
    float tA = c1 * (mv * mv + rsv * msqv * og - Bi / (RSQ * g));
    float tB = c2 * (vsq + Ci);
    float cm = tA - tB;
    float md2 = 0.f;
    for (int d = 0; d < 64; ++d) {
      float vd = xr[d] + jmlds[wave][lane * 64 + d] - alpha * mr[d];
      float tC = 2.f * c2 * (mv * vd + rsv * mr[d] * og - dlds[wave][lane * 64 + d] / (RSQ * g));
      float d2 = cm * mr[d] - tC;
      md2 += mr[d] * d2;
      dlds[wave][lane * 64 + d] = d2;                          // D consumed; reuse for d2
    }
    float ffc = (1.f + g) / (2.f * BETA_F);
    float proj = md2 / (RSQ * g * og - msqv);
    float fsq = 0.f;
    for (int d = 0; d < 64; ++d) {
      float fd = xr[d] + jmlds[wave][lane * 64 + d]
               + ffc * (dlds[wave][lane * 64 + d] + proj * mr[d]);
      dlds[wave][lane * 64 + d] = fd;                          // reuse again for f
      fsq += fd * fd;
    }
    float gf = sqrtf(1.f + (BETA_F * BETA_F) * fsq / RSQ);
    float scl = BETA_F / (1.f + gf);
    msqd[h * 1024 + i] = scl * scl * fsq;
    for (int d = 0; d < 64; ++d) {
      float mn = scl * dlds[wave][lane * 64 + d];
      m32d[base + d] = mn;
      m16d[base + d] = (_Float16)mn;
      mT16d[((size_t)h * 64 + d) * 1024 + i] = (_Float16)mn;
    }
  }
}

// ---------------- (h,n,d) -> (n, h*d) output ----------------
__global__ void k_out(const float* __restrict__ m32, float* __restrict__ out) {
  int gid = blockIdx.x * blockDim.x + threadIdx.x;   // 1M
  int i = gid >> 10, c = gid & 1023;
  int h = c >> 6, d = c & 63;
  out[gid] = m32[((size_t)h * 1024 + i) * 64 + d];
}

extern "C" void kernel_launch(void* const* d_in, const int* in_sizes, int n_in,
                              void* d_out, int out_size, void* d_ws, size_t ws_size,
                              hipStream_t stream) {
  (void)in_sizes; (void)n_in; (void)out_size; (void)ws_size;
  const float* x   = (const float*)d_in[0];
  const float* wqk = (const float*)d_in[1];
  float* out = (float*)d_out;
  char* ws = (char*)d_ws;
  size_t off = 0;
  auto carve = [&](size_t bytes) -> void* {
    void* p = ws + off;
    off = (off + bytes + 255) & ~(size_t)255;
    return p;
  };
  float*    xh    = (float*)   carve(16ull * 1024 * 64 * 4);   // normalized x (h,n,d)
  _Float16* xn16  = (_Float16*)carve(1024ull * 1024 * 2);      // normalized x (n,dim) f16
  _Float16* w16   = (_Float16*)carve(2048ull * 1024 * 2);
  _Float16* q16   = (_Float16*)carve(16ull * 1024 * 64 * 2);
  _Float16* k16   = (_Float16*)carve(16ull * 1024 * 64 * 2);
  _Float16* J16   = (_Float16*)carve(16ull * 1024 * 1024 * 2); // 32MB, L2-resident
  float*    rsb   = (float*)   carve(16ull * 1024 * 4);        // rowsum of J^2
  float*    m32a  = (float*)   carve(16ull * 1024 * 64 * 4);
  float*    m32b  = (float*)   carve(16ull * 1024 * 64 * 4);
  _Float16* m16a  = (_Float16*)carve(16ull * 1024 * 64 * 2);
  _Float16* m16b  = (_Float16*)carve(16ull * 1024 * 64 * 2);
  _Float16* mT16a = (_Float16*)carve(16ull * 1024 * 64 * 2);
  _Float16* mT16b = (_Float16*)carve(16ull * 1024 * 64 * 2);
  float*    msqa  = (float*)   carve(16ull * 1024 * 4);
  float*    msqb  = (float*)   carve(16ull * 1024 * 4);

  k_w_to_f16<<<8192, 256, 0, stream>>>(wqk, w16, 2048 * 1024);
  k_norm_init<<<64, 256, 0, stream>>>(x, xh, xn16, m32a, m16a, mT16a, msqa);
  k_qk<<<512, 128, 0, stream>>>(xn16, w16, q16, k16);
  k_attn<<<256, 128, 0, stream>>>(q16, k16, J16, rsb);
  for (int it = 1; it <= NITER; ++it) {
    if (it & 1)
      k_iter<<<256, 128, 0, stream>>>(J16, rsb, xh, m32a, m16a, mT16a, msqa,
                                      m32b, m16b, mT16b, msqb);
    else
      k_iter<<<256, 128, 0, stream>>>(J16, rsb, xh, m32b, m16b, mT16b, msqb,
                                      m32a, m16a, mT16a, msqa);
  }
  // NITER even -> final state in the 'a' set
  k_out<<<4096, 256, 0, stream>>>(m32a, out);
}
